// SecureGELUFeedForward_70214125355100
// MI455X (gfx1250) — matrix operand order repaired and checked by reading, not compiled, not run on hardware
//
#include <hip/hip_runtime.h>

// Fused secure-GELU FFN for MI455X (gfx1250, wave32).
//   Out = gelu_poly(X @ Win + bin) @ Wout + bout
//
// Round-2 design:
//  * Pre-pass splits+transposes weights ONCE into bf16 hi/lo (split-bf16 fp32
//    emulation: A*B ~= AhiBhi + AhiBlo + AloBhi on v_wmma_f32_16x16x32_bf16).
//  * Main kernel: X tile (64 rows) split-resident in LDS; B fragments stream
//    straight from L2 (weights = 32 MB << 192 MB L2) as contiguous b128 loads;
//    GEMM1 k-loop is barrier-free; intermediate GELU chunk lives in LDS only.

#define D_DIM 1024
#define F_DIM 4096
#define BM    64     // rows per block
#define BF    128    // F-chunk width
#define NT    512    // 16 wave32

typedef __attribute__((ext_vector_type(16))) __bf16       bf16x16;
typedef __attribute__((ext_vector_type(8)))  float        f32x8;
typedef __attribute__((ext_vector_type(4)))  unsigned int u32x4;

union Frag16 { bf16x16 v; u32x4 q[2]; };

__device__ __forceinline__ void splitf(float x, __bf16* hi, __bf16* lo) {
    __bf16 hb = (__bf16)x;
    *hi = hb;
    *lo = (__bf16)(x - (float)hb);
}

__device__ __forceinline__ float gelu_poly(float x) {
    float z = 1.702f * x;
    return x * (0.5f + 0.25f * z - 0.0208f * (z * z * z));
}

// ---------------- Pre-pass: split f32 [R][C] -> bf16 hi/lo transposed [C][R] ----------------
__global__ __launch_bounds__(256)
void split_transpose(const float* __restrict__ src,
                     __bf16* __restrict__ dhi, __bf16* __restrict__ dlo,
                     int R, int C) {
    __shared__ float tile[32][33];
    const int r0 = blockIdx.y * 32, c0 = blockIdx.x * 32;
    for (int i = threadIdx.x; i < 32 * 32; i += 256) {
        int r = i >> 5, c = i & 31;
        tile[r][c] = src[(size_t)(r0 + r) * C + (c0 + c)];
    }
    __syncthreads();
    for (int i = threadIdx.x; i < 32 * 32; i += 256) {
        int c = i >> 5, r = i & 31;   // consecutive threads -> consecutive r (coalesced out)
        float f = tile[r][c];
        __bf16 hi = (__bf16)f;
        dhi[(size_t)(c0 + c) * R + (r0 + r)] = hi;
        dlo[(size_t)(c0 + c) * R + (r0 + r)] = (__bf16)(f - (float)hi);
    }
}

// ---------------- Main fused FFN ----------------
__global__ __launch_bounds__(NT, 1)
void secure_gelu_ffn(const float* __restrict__ X,
                     const float* __restrict__ bin,
                     const float* __restrict__ bout,
                     const __bf16* __restrict__ WinHi,   // [F][D] transposed
                     const __bf16* __restrict__ WinLo,
                     const __bf16* __restrict__ WoutHi,  // [D][F] transposed
                     const __bf16* __restrict__ WoutLo,
                     float* __restrict__ Out)
{
    // 288 KB of 320 KB/WGP
    __shared__ __align__(16) __bf16 Xhi[BM * D_DIM];   // 128 KB
    __shared__ __align__(16) __bf16 Xlo[BM * D_DIM];   // 128 KB
    __shared__ __align__(16) __bf16 Hhi[BM * BF];      //  16 KB
    __shared__ __align__(16) __bf16 Hlo[BM * BF];      //  16 KB

    const int tid = threadIdx.x;
    const int w   = tid >> 5;     // wave 0..15
    const int l   = tid & 31;
    const int ln  = l & 15;       // lane within half
    const int hh  = l >> 4;       // lane half (K-split per WMMA layout)
    const int a   = w & 1;        // row-pair: rows a*32 .. a*32+31 (two 16-row tiles)
    const int cg  = w >> 1;       // 0..7: GEMM1 col-tile / GEMM2 col-group
    const int m0  = blockIdx.x * BM;

    // Stage + split resident X tile (once per block; i == row*D + col)
    for (int i = tid; i < BM * D_DIM; i += NT)
        splitf(X[(size_t)m0 * D_DIM + i], &Xhi[i], &Xlo[i]);
    __syncthreads();

    // Persistent GEMM2 accumulators: rows {a*32+rr*16..}, cols cg*128 + j*16 ..
    f32x8 acc2[2][8];
#pragma unroll
    for (int rr = 0; rr < 2; ++rr)
#pragma unroll
        for (int j = 0; j < 8; ++j)
#pragma unroll
            for (int e = 0; e < 8; ++e) acc2[rr][j][e] = 0.0f;

    // A-fragment element offsets in the X tile (16-bit A layout: lane half hh
    // holds K = hh*8+{0..7} and 16+hh*8+{0..7}; second run = +2 u32x4)
    const int aOff0 = (a * 32 + ln)      * D_DIM + hh * 8;
    const int aOff1 = (a * 32 + 16 + ln) * D_DIM + hh * 8;

    for (int fc = 0; fc < F_DIM; fc += BF) {
        // ---------- GEMM1: Hc = X(64x1024) @ Win[:, fc:fc+128]  (barrier-free) ----------
        f32x8 acc1[2];
#pragma unroll
        for (int rr = 0; rr < 2; ++rr)
#pragma unroll
            for (int e = 0; e < 8; ++e) acc1[rr][e] = 0.0f;

        // lane's B row: WinT[fc + cg*16 + ln][k], contiguous along K
        const __bf16* bRowHi = WinHi + (size_t)(fc + cg * 16 + ln) * D_DIM + hh * 8;
        const __bf16* bRowLo = WinLo + (size_t)(fc + cg * 16 + ln) * D_DIM + hh * 8;

#pragma unroll 4
        for (int kk = 0; kk < D_DIM; kk += 32) {
            if ((kk & 127) == 0) {                    // stream prefetch, 8 k-steps ahead
                __builtin_prefetch(bRowHi + kk + 256, 0, 0);
                __builtin_prefetch(bRowLo + kk + 256, 0, 0);
            }
            Frag16 bh, bl;
            bh.q[0] = *(const u32x4*)(bRowHi + kk);
            bh.q[1] = *(const u32x4*)(bRowHi + kk + 16);
            bl.q[0] = *(const u32x4*)(bRowLo + kk);
            bl.q[1] = *(const u32x4*)(bRowLo + kk + 16);

            Frag16 ah0, al0, ah1, al1;
            const u32x4* pa;
            pa = (const u32x4*)&Xhi[aOff0 + kk]; ah0.q[0] = pa[0]; ah0.q[1] = pa[2];
            pa = (const u32x4*)&Xlo[aOff0 + kk]; al0.q[0] = pa[0]; al0.q[1] = pa[2];
            pa = (const u32x4*)&Xhi[aOff1 + kk]; ah1.q[0] = pa[0]; ah1.q[1] = pa[2];
            pa = (const u32x4*)&Xlo[aOff1 + kk]; al1.q[0] = pa[0]; al1.q[1] = pa[2];

            acc1[0] = __builtin_amdgcn_wmma_f32_16x16x32_bf16(false, ah0.v, false, bh.v, (short)0, acc1[0], false, false);
            acc1[0] = __builtin_amdgcn_wmma_f32_16x16x32_bf16(false, ah0.v, false, bl.v, (short)0, acc1[0], false, false);
            acc1[0] = __builtin_amdgcn_wmma_f32_16x16x32_bf16(false, al0.v, false, bh.v, (short)0, acc1[0], false, false);
            acc1[1] = __builtin_amdgcn_wmma_f32_16x16x32_bf16(false, ah1.v, false, bh.v, (short)0, acc1[1], false, false);
            acc1[1] = __builtin_amdgcn_wmma_f32_16x16x32_bf16(false, ah1.v, false, bl.v, (short)0, acc1[1], false, false);
            acc1[1] = __builtin_amdgcn_wmma_f32_16x16x32_bf16(false, al1.v, false, bh.v, (short)0, acc1[1], false, false);
        }

        __syncthreads();  // previous chunk's GEMM2 readers of Hc are done

        // Bias + poly-GELU + split-store chunk (C layout: VGPR i -> M=i+8*hh, N=ln)
        {
            float bias = bin[fc + cg * 16 + ln];
#pragma unroll
            for (int rr = 0; rr < 2; ++rr)
#pragma unroll
                for (int i2 = 0; i2 < 8; ++i2) {
                    float v = gelu_poly(acc1[rr][i2] + bias);
                    int m = a * 32 + rr * 16 + i2 + 8 * hh;
                    int c = cg * 16 + ln;
                    splitf(v, &Hhi[m * BF + c], &Hlo[m * BF + c]);
                }
        }
        __syncthreads();  // Hc visible to all waves

        // ---------- GEMM2: acc2 += Hc(64x128) @ Wout[fc:fc+128, :] ----------
#pragma unroll
        for (int k2 = 0; k2 < BF; k2 += 32) {
            Frag16 ah0, al0, ah1, al1;
            const u32x4* pa;
            pa = (const u32x4*)&Hhi[(a * 32 + ln)      * BF + k2 + hh * 8]; ah0.q[0] = pa[0]; ah0.q[1] = pa[2];
            pa = (const u32x4*)&Hlo[(a * 32 + ln)      * BF + k2 + hh * 8]; al0.q[0] = pa[0]; al0.q[1] = pa[2];
            pa = (const u32x4*)&Hhi[(a * 32 + 16 + ln) * BF + k2 + hh * 8]; ah1.q[0] = pa[0]; ah1.q[1] = pa[2];
            pa = (const u32x4*)&Hlo[(a * 32 + 16 + ln) * BF + k2 + hh * 8]; al1.q[0] = pa[0]; al1.q[1] = pa[2];

#pragma unroll
            for (int j = 0; j < 8; ++j) {
                const __bf16* rHi = WoutHi + (size_t)(cg * 128 + j * 16 + ln) * F_DIM + fc + k2 + hh * 8;
                const __bf16* rLo = WoutLo + (size_t)(cg * 128 + j * 16 + ln) * F_DIM + fc + k2 + hh * 8;
                if (k2 == 0) {                       // next-chunk stream prefetch
                    __builtin_prefetch(rHi + BF, 0, 0);
                    __builtin_prefetch(rLo + BF, 0, 0);
                }
                Frag16 bh, bl;
                bh.q[0] = *(const u32x4*)(rHi);
                bh.q[1] = *(const u32x4*)(rHi + 16);
                bl.q[0] = *(const u32x4*)(rLo);
                bl.q[1] = *(const u32x4*)(rLo + 16);

                acc2[0][j] = __builtin_amdgcn_wmma_f32_16x16x32_bf16(false, ah0.v, false, bh.v, (short)0, acc2[0][j], false, false);
                acc2[0][j] = __builtin_amdgcn_wmma_f32_16x16x32_bf16(false, ah0.v, false, bl.v, (short)0, acc2[0][j], false, false);
                acc2[0][j] = __builtin_amdgcn_wmma_f32_16x16x32_bf16(false, al0.v, false, bh.v, (short)0, acc2[0][j], false, false);
                acc2[1][j] = __builtin_amdgcn_wmma_f32_16x16x32_bf16(false, ah1.v, false, bh.v, (short)0, acc2[1][j], false, false);
                acc2[1][j] = __builtin_amdgcn_wmma_f32_16x16x32_bf16(false, ah1.v, false, bl.v, (short)0, acc2[1][j], false, false);
                acc2[1][j] = __builtin_amdgcn_wmma_f32_16x16x32_bf16(false, al1.v, false, bh.v, (short)0, acc2[1][j], false, false);
            }
        }
    }

    // Epilogue: bias + store
#pragma unroll
    for (int j = 0; j < 8; ++j) {
        int n = cg * 128 + j * 16 + ln;
        float bo = bout[n];
#pragma unroll
        for (int rr = 0; rr < 2; ++rr)
#pragma unroll
            for (int i2 = 0; i2 < 8; ++i2) {
                int m = a * 32 + rr * 16 + i2 + 8 * hh;
                Out[(size_t)(m0 + m) * D_DIM + n] = acc2[rr][j][i2] + bo;
            }
    }
}

extern "C" void kernel_launch(void* const* d_in, const int* in_sizes, int n_in,
                              void* d_out, int out_size, void* d_ws, size_t ws_size,
                              hipStream_t stream) {
    const float* X    = (const float*)d_in[0];   // (B,S,D) f32
    const float* Win  = (const float*)d_in[1];   // (D,F)
    const float* bin  = (const float*)d_in[2];   // (F)
    const float* Wout = (const float*)d_in[3];   // (F,D)
    const float* bout = (const float*)d_in[4];   // (D)
    float* Out = (float*)d_out;

    // Workspace: 4 x (4M bf16) = 32 MB of pre-split transposed weights
    __bf16* winHi  = (__bf16*)d_ws;
    __bf16* winLo  = winHi  + (size_t)F_DIM * D_DIM;
    __bf16* woutHi = winLo  + (size_t)F_DIM * D_DIM;
    __bf16* woutLo = woutHi + (size_t)D_DIM * F_DIM;

    // Pre-pass: Win [D][F] -> WinT [F][D]; Wout [F][D] -> WoutT [D][F]
    hipLaunchKernelGGL(split_transpose, dim3(F_DIM / 32, D_DIM / 32), dim3(256), 0, stream,
                       Win, winHi, winLo, D_DIM, F_DIM);
    hipLaunchKernelGGL(split_transpose, dim3(D_DIM / 32, F_DIM / 32), dim3(256), 0, stream,
                       Wout, woutHi, woutLo, F_DIM, D_DIM);

    int M = in_sizes[0] / D_DIM;                 // 16384 rows
    hipLaunchKernelGGL(secure_gelu_ffn, dim3(M / BM), dim3(NT), 0, stream,
                       X, bin, bout, winHi, winLo, woutHi, woutLo, Out);
}
